// Decoder_85298050498796
// MI455X (gfx1250) — matrix operand order repaired and checked by reading, not compiled, not run on hardware
//
#include <hip/hip_runtime.h>
#include <hip/hip_bf16.h>

// ---------------- problem constants ----------------
#define BB   64
#define SS   1024
#define HH   1024
#define DD   2048
#define EMBD 512
#define H3   3072

typedef __attribute__((ext_vector_type(16))) __bf16 v16bf;
typedef __attribute__((ext_vector_type(8)))  __bf16 v8bf;
typedef __attribute__((ext_vector_type(8)))  float  v8f;

static __device__ inline __bf16 f2bf(float f) { return (__bf16)f; }

// ---------------- embedding gather -> bf16: y_bf[b,:] = bf16(emb[prev_y[b],:]) ------
__global__ void embed_bf_kernel(const int* __restrict__ prev_y,
                                const float* __restrict__ emb,
                                __bf16* __restrict__ y_bf) {
    const int b = blockIdx.x;
    const long long row = prev_y[b];
    const float* src = emb + row * (long long)EMBD;
    __bf16* dst = y_bf + (long long)b * EMBD;
    for (int j = threadIdx.x; j < EMBD; j += blockDim.x) dst[j] = f2bf(src[j]);
}

// ---------------- f32 -> bf16 pack ----------------
__global__ void pack_bf16_kernel(const float* __restrict__ src,
                                 __bf16* __restrict__ dst, int n) {
    const int i = blockIdx.x * blockDim.x + threadIdx.x;
    if (i < n) dst[i] = f2bf(src[i]);
}

// ---------------- WMMA GEMM: Out[64 x N] = Xbf[64 x K] * W[K x N] ----------------
// blockDim = 128 (4 wave32s). Wave w owns M-tile w; blockIdx.x owns a 16-col N tile.
// A is pre-packed bf16 row-major: the 16x32 A fragment is two contiguous 16B loads
// per lane. B (weights, f32, single-use) is converted inline with native cvt.
__global__ __launch_bounds__(128) void gemm_wmma_kernel(
        const __bf16* __restrict__ Xbf, const float* __restrict__ W,
        float* __restrict__ Out, int K, int N) {
    const int lane  = threadIdx.x & 31;
    const int wave  = threadIdx.x >> 5;
    const int m0    = wave * 16;
    const int n0    = blockIdx.x * 16;
    const int mrow  = m0 + (lane & 15);
    const int khalf = (lane >> 4) ? 8 : 0;          // lanes 16..31 hold K+8 pairs

    v8f acc = {};
    const __bf16* xrow = Xbf + (long long)mrow * K;

    for (int k0 = 0; k0 < K; k0 += 32) {
        // A fragment: halves [k0+khalf .. +7] and [k0+16+khalf .. +7] (ISA 7.12.2)
        const v8bf alo = *(const v8bf*)(xrow + k0 + khalf);
        const v8bf ahi = *(const v8bf*)(xrow + k0 + 16 + khalf);
        const v16bf a = __builtin_shufflevector(alo, ahi,
                0, 1, 2, 3, 4, 5, 6, 7, 8, 9, 10, 11, 12, 13, 14, 15);

        // B fragment: lane = N col (n0+lane&15), K rows [k0+khalf .. +7] and [k0+16+khalf .. +7] -> bf16
        const float* wcol = W + (long long)(k0 + khalf) * N + n0 + (lane & 15);
        v16bf b;
        #pragma unroll
        for (int i = 0; i < 8; ++i) {
            b[i]     = f2bf(wcol[(long long)i * N]);
            b[8 + i] = f2bf(wcol[(long long)(16 + i) * N]);
        }
        acc = __builtin_amdgcn_wmma_f32_16x16x32_bf16(
                  false, a, false, b, (short)0, acc, false, false);
    }

    // C/D layout: VGPR r -> M = r (lanes 0..15) or r+8 (lanes 16..31), N = lane&15
    const int rbase = m0 + ((lane >> 4) ? 8 : 0);
    const int ocol  = n0 + (lane & 15);
    #pragma unroll
    for (int r = 0; r < 8; ++r)
        Out[(long long)(rbase + r) * N + ocol] = acc[r];
}

// ---------------- fused LayerNorm(ax)+LayerNorm(ah)+GRU gates ----------------
// one block per batch row; 256 threads; LN over the full 3H=3072 vector.
// Also emits a bf16 copy of the new hidden state for the next WMMA GEMM.
__global__ __launch_bounds__(256) void lngru_gates_kernel(
        const float* __restrict__ ax_raw, const float* __restrict__ ah_raw,
        const float* __restrict__ gx, const float* __restrict__ bx,
        const float* __restrict__ gh, const float* __restrict__ bh,
        const float* __restrict__ h_in, float* __restrict__ h_out,
        __bf16* __restrict__ hb_out) {
    const int b   = blockIdx.x;
    const int tid = threadIdx.x;
    const float* axp = ax_raw + (long long)b * H3;
    const float* ahp = ah_raw + (long long)b * H3;

    __shared__ float r0[256], r1[256], r2[256], r3[256];
    float sx = 0.f, sxx = 0.f, sh = 0.f, shh = 0.f;
    for (int j = tid; j < H3; j += 256) {
        const float vx = axp[j], vh = ahp[j];
        sx += vx; sxx += vx * vx; sh += vh; shh += vh * vh;
    }
    r0[tid] = sx; r1[tid] = sxx; r2[tid] = sh; r3[tid] = shh;
    __syncthreads();
    for (int off = 128; off > 0; off >>= 1) {
        if (tid < off) {
            r0[tid] += r0[tid + off]; r1[tid] += r1[tid + off];
            r2[tid] += r2[tid + off]; r3[tid] += r3[tid + off];
        }
        __syncthreads();
    }
    const float inv3h = 1.f / (float)H3;
    const float mux = r0[0] * inv3h;
    const float muh = r2[0] * inv3h;
    const float rsx = rsqrtf(fmaxf(r1[0] * inv3h - mux * mux, 0.f) + 1e-5f);
    const float rsh = rsqrtf(fmaxf(r3[0] * inv3h - muh * muh, 0.f) + 1e-5f);

    const float* hin  = h_in  + (long long)b * HH;
    float*       hout = h_out + (long long)b * HH;
    for (int j = tid; j < HH; j += 256) {
        const float axr = (axp[j]          - mux) * rsx * gx[j]          + bx[j];
        const float axz = (axp[j + HH]     - mux) * rsx * gx[j + HH]     + bx[j + HH];
        const float axn = (axp[j + 2 * HH] - mux) * rsx * gx[j + 2 * HH] + bx[j + 2 * HH];
        const float ahr = (ahp[j]          - muh) * rsh * gh[j]          + bh[j];
        const float ahz = (ahp[j + HH]     - muh) * rsh * gh[j + HH]     + bh[j + HH];
        const float ahn = (ahp[j + 2 * HH] - muh) * rsh * gh[j + 2 * HH] + bh[j + 2 * HH];
        const float r = 1.f / (1.f + __expf(-(axr + ahr)));
        const float z = 1.f / (1.f + __expf(-(axz + ahz)));
        const float n = tanhf(axn + r * ahn);
        const float h = (1.f - z) * n + z * hin[j];
        hout[j] = h;
        if (hb_out) hb_out[(long long)b * HH + j] = f2bf(h);
    }
}

// ---------------- attention scores: scores[b,s] = <ctx[b,s,:], q[b,:]> ----------------
__global__ __launch_bounds__(256) void attn_scores_kernel(
        const float* __restrict__ ctx, const float* __restrict__ q,
        const unsigned char* __restrict__ mask, float* __restrict__ scores) {
    const int b = blockIdx.y;
    const int s = blockIdx.x * 256 + threadIdx.x;
    const float4* row = (const float4*)(ctx + ((long long)b * SS + s) * DD);
    const float4* qv  = (const float4*)(q + (long long)b * DD);
    float acc = 0.f;
    #pragma unroll 4
    for (int i = 0; i < DD / 4; ++i) {
        const float4 c = row[i], d = qv[i];
        acc += c.x * d.x + c.y * d.y + c.z * d.z + c.w * d.w;
    }
    if (!mask[(long long)b * SS + s]) acc = -3.402823466e38f;
    scores[(long long)b * SS + s] = acc;
}

// ---------------- softmax over S per batch row; writes At ----------------
__global__ __launch_bounds__(256) void attn_softmax_kernel(
        const float* __restrict__ scores, float* __restrict__ At) {
    const int b = blockIdx.x, tid = threadIdx.x;
    __shared__ float red[256];
    float v[SS / 256];
    float mx = -3.402823466e38f;
    #pragma unroll
    for (int j = 0; j < SS / 256; ++j) {
        v[j] = scores[(long long)b * SS + tid + j * 256];
        mx = fmaxf(mx, v[j]);
    }
    red[tid] = mx; __syncthreads();
    for (int off = 128; off > 0; off >>= 1) {
        if (tid < off) red[tid] = fmaxf(red[tid], red[tid + off]);
        __syncthreads();
    }
    mx = red[0]; __syncthreads();
    float sum = 0.f;
    #pragma unroll
    for (int j = 0; j < SS / 256; ++j) { v[j] = __expf(v[j] - mx); sum += v[j]; }
    red[tid] = sum; __syncthreads();
    for (int off = 128; off > 0; off >>= 1) {
        if (tid < off) red[tid] += red[tid + off];
        __syncthreads();
    }
    const float inv = 1.f / red[0];
    #pragma unroll
    for (int j = 0; j < SS / 256; ++j)
        At[(long long)b * SS + tid + j * 256] = v[j] * inv;
}

// ------- Ct_bf[b,d] = bf16( sum_s At[b,s] * ctx[b,s,d] ) (coalesced over d) ---------
__global__ __launch_bounds__(256) void attn_ct_kernel(
        const float* __restrict__ At, const float* __restrict__ ctx,
        __bf16* __restrict__ Ct_bf) {
    const int b = blockIdx.y;
    const int d = blockIdx.x * 256 + threadIdx.x;
    __shared__ float a[SS];
    for (int j = threadIdx.x; j < SS; j += 256) a[j] = At[(long long)b * SS + j];
    __syncthreads();
    const float* base = ctx + (long long)b * SS * DD + d;
    float acc = 0.f;
    #pragma unroll 4
    for (int s = 0; s < SS; ++s) acc += a[s] * base[(long long)s * DD];
    Ct_bf[(long long)b * DD + d] = f2bf(acc);
}

// ---------------- helpers ----------------
__global__ void zero_kernel(float* __restrict__ p, int n) {
    const int i = blockIdx.x * blockDim.x + threadIdx.x;
    if (i < n) p[i] = 0.f;
}
__global__ void dup_h_kernel(const float* __restrict__ h, float* __restrict__ out) {
    const int i = blockIdx.x * blockDim.x + threadIdx.x;   // n = B*H
    const float v = h[i];
    out[i] = v;
    out[BB * HH + i] = v;
}

// ---------------- launch ----------------
extern "C" void kernel_launch(void* const* d_in, const int* in_sizes, int n_in,
                              void* d_out, int out_size, void* d_ws, size_t ws_size,
                              hipStream_t stream) {
    (void)in_sizes; (void)n_in; (void)out_size; (void)ws_size;
    const unsigned char* mask = (const unsigned char*)d_in[0];
    const float* ctx    = (const float*)d_in[1];
    const int*   prev_y = (const int*)  d_in[2];
    const float* hiddens= (const float*)d_in[3];
    const float* emb    = (const float*)d_in[4];
    const float* Wa     = (const float*)d_in[5];
    const float* Wx0    = (const float*)d_in[6];
    const float* Wh0    = (const float*)d_in[7];
    const float* gx0 = (const float*)d_in[8],  *bx0 = (const float*)d_in[9];
    const float* gh0 = (const float*)d_in[10], *bh0 = (const float*)d_in[11];
    const float* Wx1    = (const float*)d_in[12];
    const float* Wh1    = (const float*)d_in[13];
    const float* gx1 = (const float*)d_in[14], *bx1 = (const float*)d_in[15];
    const float* gh1 = (const float*)d_in[16], *bh1 = (const float*)d_in[17];
    // d_in[18] = Wx_e (unused: zero input makes x@Wx exactly 0)
    const float* Wh_e = (const float*)d_in[19];
    const float* gx_e = (const float*)d_in[20], *bx_e = (const float*)d_in[21];
    const float* gh_e = (const float*)d_in[22], *bh_e = (const float*)d_in[23];

    // ---- workspace layout: f32 region first, then 16B-aligned bf16 region ----
    float* ws = (float*)d_ws;
    float* ax = ws;                         // 64*3072
    float* ah = ax + BB * H3;               // 64*3072
    float* h0 = ah + BB * H3;               // 64*1024
    float* h1 = h0 + BB * HH;
    float* h2 = h1 + BB * HH;
    float* h3 = h2 + BB * HH;
    float* q  = h3 + BB * HH;               // 64*2048
    float* sc = q  + BB * DD;               // 64*1024
    __bf16* bfbase = (__bf16*)(sc + BB * SS);
    __bf16* y_bf   = bfbase;                // 64*512
    __bf16* hid_bf = y_bf   + BB * EMBD;    // 64*1024
    __bf16* h0_bf  = hid_bf + BB * HH;
    __bf16* h1_bf  = h0_bf  + BB * HH;
    __bf16* h2_bf  = h1_bf  + BB * HH;
    __bf16* Ct_bf  = h2_bf  + BB * HH;      // 64*2048

    float* out = (float*)d_out;
    float* At  = out + 2 * BB * HH;         // third output slot

    // ---- GRU cell 0 ----
    embed_bf_kernel<<<BB, 256, 0, stream>>>(prev_y, emb, y_bf);
    pack_bf16_kernel<<<(BB * HH) / 256, 256, 0, stream>>>(hiddens, hid_bf, BB * HH);
    gemm_wmma_kernel<<<H3 / 16, 128, 0, stream>>>(y_bf,   Wx0, ax, EMBD, H3);
    gemm_wmma_kernel<<<H3 / 16, 128, 0, stream>>>(hid_bf, Wh0, ah, HH,   H3);
    lngru_gates_kernel<<<BB, 256, 0, stream>>>(ax, ah, gx0, bx0, gh0, bh0,
                                               hiddens, h0, h0_bf);

    // ---- attention ----
    gemm_wmma_kernel<<<DD / 16, 128, 0, stream>>>(h0_bf, Wa, q, HH, DD);
    attn_scores_kernel<<<dim3(SS / 256, BB), 256, 0, stream>>>(ctx, q, mask, sc);
    attn_softmax_kernel<<<BB, 256, 0, stream>>>(sc, At);
    attn_ct_kernel<<<dim3(DD / 256, BB), 256, 0, stream>>>(At, ctx, Ct_bf);

    // ---- GRU cell 1 ----
    gemm_wmma_kernel<<<H3 / 16, 128, 0, stream>>>(Ct_bf, Wx1, ax, DD, H3);
    gemm_wmma_kernel<<<H3 / 16, 128, 0, stream>>>(h0_bf, Wh1, ah, HH, H3);
    lngru_gates_kernel<<<BB, 256, 0, stream>>>(ax, ah, gx1, bx1, gh1, bh1,
                                               h0, h1, h1_bf);

    // ---- tail GRU layers (zero input: LN(0)*g + b == b; feed zeroed ax_raw) ----
    zero_kernel<<<(BB * H3 + 255) / 256, 256, 0, stream>>>(ax, BB * H3);

    gemm_wmma_kernel<<<H3 / 16, 128, 0, stream>>>(h1_bf, Wh_e, ah, HH, H3);
    lngru_gates_kernel<<<BB, 256, 0, stream>>>(ax, ah, gx_e, bx_e, gh_e, bh_e,
                                               h1, h2, h2_bf);

    gemm_wmma_kernel<<<H3 / 16, 128, 0, stream>>>(h2_bf, Wh_e + (long long)HH * H3,
                                                  ah, HH, H3);
    lngru_gates_kernel<<<BB, 256, 0, stream>>>(ax, ah, gx_e + H3, bx_e + H3,
                                               gh_e + H3, bh_e + H3, h2, h3, nullptr);

    // ---- outputs: (h, h, At) ----
    dup_h_kernel<<<(BB * HH) / 256, 256, 0, stream>>>(h3, out);
}
